// SparseNeighborAttention_11390253269522
// MI455X (gfx1250) — matrix operand order repaired and checked by reading, not compiled, not run on hardware
//
#include <hip/hip_runtime.h>
#include <hip/hip_bf16.h>

// Sparse neighbor attention for MI455X (gfx1250, wave32, WMMA).
// Strategy: all GEMM operands in bf16 (halves the L2-resident neighbor
// gather traffic; K/V tables = 25.6MB each, fully L2-resident in 192MB L2),
// fp32 accumulation via v_wmma_f32_16x16x32_bf16.

#define HID  256
#define NHEAD 8
#define DH   32
#define KNBR 64

typedef __attribute__((ext_vector_type(16))) __bf16 v16bf;
typedef __attribute__((ext_vector_type(8)))  float  v8f;

union frag_cvt { uint4 u[2]; v16bf v; };
union u4bf     { uint4 u;    __bf16 h[8]; };

// ---------------------------------------------------------------- convert
__global__ void cvt_f32_to_bf16(const float* __restrict__ src,
                                __bf16* __restrict__ dst, int n) {
  int i = blockIdx.x * blockDim.x + threadIdx.x;
  if (i < n) dst[i] = (__bf16)src[i];
}

// ---------------------------------------------------------------- GEMM
// C[M,256] = A[M,256] @ W[256,256]^T (+bias).  W row-major W[j][i] is the
// natural contiguous layout for the B-fragment of y = x @ W^T.
// Block = 256 thr = 8 waves; wave w -> rows [blk.x*128 + 16w, +16),
// cols [blk.y*64, +64) as four 16x16 WMMA tiles. 32 WMMAs per wave.
__global__ __launch_bounds__(256) void gemm_bf16_wmma(
    const __bf16* __restrict__ A, const __bf16* __restrict__ W,
    void* __restrict__ out, const float* __restrict__ bias,
    int M, int f32out) {
  const int lane  = threadIdx.x & 31;
  const int wave  = threadIdx.x >> 5;
  const int m0    = blockIdx.x * 128 + wave * 16;
  const int n0    = blockIdx.y * 64;

  // A-fragment lane mapping (16-bit A 16x32): lanes 0-15 hold K {0..7,16..23}
  // of row M=lane; lanes 16-31 hold K {8..15,24..31} of row M=lane-16.
  const int mrow  = min(m0 + (lane & 15), M - 1);   // clamp OOB rows (loads only)
  const int akb   = (lane < 16) ? 0 : 8;
  // B-fragment lane mapping (16-bit B 32x16): lanes 0-15 hold K {0..7,16..23} of
  // col N=lane; lanes 16-31 hold K {8..15,24..31} of col N=lane-16.
  const int bkoff = (lane < 16) ? 0 : 8;
  const int ncol  = lane & 15;

  v8f c[4];
#pragma unroll
  for (int s = 0; s < 4; ++s) c[s] = {};

#pragma unroll
  for (int kk = 0; kk < HID; kk += 32) {
    const __bf16* ap = A + (size_t)mrow * HID + kk + akb;
    frag_cvt fa;
    fa.u[0] = *reinterpret_cast<const uint4*>(ap);        // K base..base+7
    fa.u[1] = *reinterpret_cast<const uint4*>(ap + 16);   // K base+16..base+23
#pragma unroll
    for (int s = 0; s < 4; ++s) {
      const __bf16* bp = W + (size_t)(n0 + s * 16 + ncol) * HID + kk + bkoff;
      frag_cvt fb;
      fb.u[0] = *reinterpret_cast<const uint4*>(bp);      // K koff..koff+7
      fb.u[1] = *reinterpret_cast<const uint4*>(bp + 16); // K koff+16..koff+23
      c[s] = __builtin_amdgcn_wmma_f32_16x16x32_bf16(
          false, fa.v, false, fb.v, (short)0, c[s], false, false);
    }
  }

  // C/D layout: VGPR i -> M = i + (lane<16 ? 0 : 8), N = n-tile + (lane&15)
  const int rbase = m0 + ((lane >> 4) << 3);
#pragma unroll
  for (int s = 0; s < 4; ++s) {
    const int n  = n0 + s * 16 + ncol;
    const float bv = bias ? bias[n] : 0.0f;
#pragma unroll
    for (int i = 0; i < 8; ++i) {
      const int row = rbase + i;
      if (row < M) {
        float val = c[s][i] + bv;
        if (f32out) ((float*)out)[(size_t)row * HID + n] = val;
        else        ((__bf16*)out)[(size_t)row * HID + n] = (__bf16)val;
      }
    }
  }
}

// ---------------------------------------------------------------- attention
// One block per node; wave h handles head h (wave32-native).
// Phase 1: lane j owns neighbors j and j+32 -> scores (per-lane 64B bf16
//          gathers, L2-resident), shfl-based softmax across 32 lanes.
// Phase 2: lane = dim -> coalesced weighted V accumulation.
__global__ __launch_bounds__(256) void neighbor_attn(
    const __bf16* __restrict__ qb, const __bf16* __restrict__ kb,
    const __bf16* __restrict__ vb, const int* __restrict__ nidx,
    const unsigned char* __restrict__ nmask,   // jnp bool -> 1 byte/elem
    __bf16* __restrict__ attn, int N) {
  const int n    = blockIdx.x;
  if (n >= N) return;
  const int lane = threadIdx.x & 31;
  const int h    = threadIdx.x >> 5;

  __shared__ int   s_idx[KNBR];
  __shared__ int   s_msk[KNBR];
  __shared__ float s_q[HID];
  __shared__ float s_w[NHEAD][KNBR];

  if (threadIdx.x < KNBR) {
    s_idx[threadIdx.x] = nidx[(size_t)n * KNBR + threadIdx.x];
    s_msk[threadIdx.x] = nmask[(size_t)n * KNBR + threadIdx.x];
  }
  s_q[threadIdx.x] = (float)qb[(size_t)n * HID + threadIdx.x];
  __syncthreads();

  const float* qh = &s_q[h * DH];
  const float scale = 0.17677669529663687f;   // 1/sqrt(D), D=32

  float sc[2];
#pragma unroll
  for (int t = 0; t < 2; ++t) {
    const int j = lane + t * 32;
    float s;
    if (s_msk[j]) {
      s = -INFINITY;
    } else {
      const uint4* kp = reinterpret_cast<const uint4*>(
          kb + (size_t)s_idx[j] * HID + h * DH);
      float acc = 0.0f;
#pragma unroll
      for (int q4 = 0; q4 < 4; ++q4) {
        u4bf cv; cv.u = kp[q4];
#pragma unroll
        for (int d = 0; d < 8; ++d) acc += qh[q4 * 8 + d] * (float)cv.h[d];
      }
      s = acc * scale;
    }
    sc[t] = s;
  }

  float mx = fmaxf(sc[0], sc[1]);
#pragma unroll
  for (int o = 16; o > 0; o >>= 1) mx = fmaxf(mx, __shfl_xor(mx, o, 32));
  const float p0 = (sc[0] == -INFINITY) ? 0.0f : __expf(sc[0] - mx);
  const float p1 = (sc[1] == -INFINITY) ? 0.0f : __expf(sc[1] - mx);
  float sum = p0 + p1;
#pragma unroll
  for (int o = 16; o > 0; o >>= 1) sum += __shfl_xor(sum, o, 32);
  const float inv = (sum > 0.0f) ? (1.0f / sum) : 0.0f;  // all-PAD -> zeros
  s_w[h][lane]      = p0 * inv;
  s_w[h][lane + 32] = p1 * inv;
  // per-wave LDS accesses are in-order; no block barrier needed (own head only)

  float acc = 0.0f;
#pragma unroll 4
  for (int j = 0; j < KNBR; ++j) {
    const float w = s_w[h][j];
    if (w != 0.0f)
      acc += w * (float)vb[(size_t)s_idx[j] * HID + h * DH + lane];
  }
  attn[(size_t)n * HID + h * DH + lane] = (__bf16)acc;
}

// ---------------------------------------------------------------- launch
extern "C" void kernel_launch(void* const* d_in, const int* in_sizes, int n_in,
                              void* d_out, int out_size, void* d_ws, size_t ws_size,
                              hipStream_t stream) {
  (void)n_in; (void)out_size; (void)ws_size;
  const float*         x  = (const float*)d_in[0];
  const int*           ni = (const int*)d_in[1];
  const unsigned char* nm = (const unsigned char*)d_in[2];
  const float*         Wq = (const float*)d_in[3];
  const float*         Wk = (const float*)d_in[4];
  const float*         Wv = (const float*)d_in[5];
  const float*         Wo = (const float*)d_in[6];
  const float*         bo = (const float*)d_in[7];

  const int N = in_sizes[0] / HID;

  char* base = (char*)d_ws;
  const size_t sz = (size_t)N * HID * sizeof(__bf16);
  __bf16* xbf = (__bf16*)(base);
  __bf16* qbf = (__bf16*)(base + sz);
  __bf16* kbf = (__bf16*)(base + 2 * sz);
  __bf16* vbf = (__bf16*)(base + 3 * sz);
  __bf16* wbf = (__bf16*)(base + 4 * sz);
  __bf16* wq = wbf, * wk = wbf + 65536, * wv = wbf + 131072, * wo = wbf + 196608;
  __bf16* attnbf = xbf;   // alias: x_bf dead after V projection

  const int totx = N * HID;
  cvt_f32_to_bf16<<<(totx + 255) / 256, 256, 0, stream>>>(x,  xbf, totx);
  cvt_f32_to_bf16<<<256, 256, 0, stream>>>(Wq, wq, 65536);
  cvt_f32_to_bf16<<<256, 256, 0, stream>>>(Wk, wk, 65536);
  cvt_f32_to_bf16<<<256, 256, 0, stream>>>(Wv, wv, 65536);
  cvt_f32_to_bf16<<<256, 256, 0, stream>>>(Wo, wo, 65536);

  dim3 g((N + 127) / 128, 4);
  gemm_bf16_wmma<<<g, 256, 0, stream>>>(xbf, wq, (void*)qbf, nullptr, N, 0);
  gemm_bf16_wmma<<<g, 256, 0, stream>>>(xbf, wk, (void*)kbf, nullptr, N, 0);
  gemm_bf16_wmma<<<g, 256, 0, stream>>>(xbf, wv, (void*)vbf, nullptr, N, 0);

  neighbor_attn<<<N, 256, 0, stream>>>(qbf, kbf, vbf, ni, nm, attnbf, N);

  gemm_bf16_wmma<<<g, 256, 0, stream>>>(attnbf, wo, d_out, bo, N, 1);
}
